// MultiHeadAttention_32882269618690
// MI455X (gfx1250) — matrix operand order repaired and checked by reading, not compiled, not run on hardware
//
#include <hip/hip_runtime.h>
#include <hip/hip_bf16.h>

// ---------------------------------------------------------------------------
// MHA forward for B=2, S=2048, E=1024, H=16, D=64 (fp32 in/out).
// bf16 WMMA (v_wmma_f32_16x16x32_bf16) everywhere, flash-attention softmax,
// async global->LDS staging (ASYNCcnt) with double buffering in the GEMM.
// ---------------------------------------------------------------------------

typedef __attribute__((ext_vector_type(16))) __bf16 bf16x16;
typedef __attribute__((ext_vector_type(8)))  float  f32x8;
typedef __attribute__((ext_vector_type(4)))  int    i32x4;

union FragB {
    bf16x16 v;
    uint4   u[2];
};

__device__ __forceinline__ f32x8 wmma_bf16(bf16x16 a, bf16x16 b, f32x8 c) {
    return __builtin_amdgcn_wmma_f32_16x16x32_bf16(
        /*neg_a=*/false, a, /*neg_b=*/false, b,
        /*c_mod=*/(short)0, c, /*reuse_a=*/false, /*reuse_b=*/false);
}

// ---- CDNA5 async global->LDS copy (ASYNCcnt path) -------------------------
#if defined(__has_builtin)
#if __has_builtin(__builtin_amdgcn_global_load_async_to_lds_b128) && \
    __has_builtin(__builtin_amdgcn_s_wait_asynccnt)
#define USE_ASYNC 1
#endif
#endif
#ifndef USE_ASYNC
#define USE_ASYNC 0
#pragma message("CDNA5 async-to-LDS builtins NOT available; using sync LDS staging")
#endif

#if USE_ASYNC
typedef __attribute__((address_space(1))) i32x4 gi32x4;   // global int4
typedef __attribute__((address_space(3))) i32x4 li32x4;   // LDS int4

__device__ __forceinline__ void async_b128(void* lds, const void* gptr) {
    __builtin_amdgcn_global_load_async_to_lds_b128(
        (gi32x4*)(gptr),
        (li32x4*)(lds),
        /*offset=*/0, /*cpol=*/0);
}
__device__ __forceinline__ void async_wait0() {
    __builtin_amdgcn_s_wait_asynccnt(0);
}
#else
__device__ __forceinline__ void async_wait0() {}
#endif

constexpr int Bd = 2;
constexpr int Sd = 2048;
constexpr int Ed = 1024;
constexpr int Hd = 16;
constexpr int Dd = 64;
constexpr int Md = Bd * Sd;   // 4096 rows

// ---------------------------------------------------------------------------
// f32 -> bf16 elementwise (x)
// ---------------------------------------------------------------------------
__global__ __launch_bounds__(256) void cvt_f32_bf16(
    const float* __restrict__ in, __bf16* __restrict__ out, int n)
{
    int i = (blockIdx.x * 256 + threadIdx.x) * 4;
    if (i + 3 < n) {
        float4 v = *(const float4*)(in + i);
        out[i + 0] = (__bf16)v.x;
        out[i + 1] = (__bf16)v.y;
        out[i + 2] = (__bf16)v.z;
        out[i + 3] = (__bf16)v.w;
    }
}

// ---------------------------------------------------------------------------
// W [K][N] f32  ->  Wt [N][K] bf16   (E = 1024 hardcoded via shifts)
// ---------------------------------------------------------------------------
__global__ __launch_bounds__(256) void transpose_cvt(
    const float* __restrict__ W, __bf16* __restrict__ Wt)
{
    int idx  = blockIdx.x * 256 + threadIdx.x;   // over E*E
    int nrow = idx >> 10;
    int kcol = idx & 1023;
    Wt[(size_t)nrow * Ed + kcol] = (__bf16)W[(size_t)kcol * Ed + nrow];
}

// ---------------------------------------------------------------------------
// Tiled bf16 WMMA GEMM:  C[M][N] = A[M][K] * Bt[N][K]^T + bias
// Block tile 128x64, BK=32, 256 threads = 8 waves, wave tile 32x32.
// Double-buffered LDS staging via async global->LDS copies.
// MODE 0: scatter bf16 to [B,H,S,D] (QKV).  MODE 1: plain f32 [M][N] (proj).
// ---------------------------------------------------------------------------
template <int MODE>
__global__ __launch_bounds__(256) void gemm_bf16(
    const __bf16* __restrict__ A, const __bf16* __restrict__ Bt,
    const float* __restrict__ bias, void* __restrict__ outv,
    int Kdim, int Ndim)
{
    __shared__ __bf16 As[2][128 * 32];   // [row][k]  2 x 8 KB
    __shared__ __bf16 Bs[2][64 * 32];    // [col][k]  2 x 4 KB

    const int t    = threadIdx.x;
    const int lane = t & 31;
    const int w    = t >> 5;
    const int wm   = w >> 1;          // 0..3
    const int wn   = w & 1;           // 0..1
    const int lh   = lane >> 4;       // half-wave select
    const int lr   = lane & 15;
    const int m0   = blockIdx.y * 128;
    const int n0   = blockIdx.x * 64;

    f32x8 acc[2][2] = {};

    const int ar = t >> 1;            // 0..127
    const int ac = (t & 1) * 16;      // 0 / 16
    const int br = t >> 2;            // 0..63
    const int bc = (t & 3) * 8;       // 0..24

    auto stage = [&](int p, int k0) {
        const __bf16* ap = A + (size_t)(m0 + ar) * Kdim + k0 + ac;
        __bf16*       la = &As[p][ar * 32 + ac];
        const __bf16* bp = Bt + (size_t)(n0 + br) * Kdim + k0 + bc;
        __bf16*       lb = &Bs[p][br * 32 + bc];
#if USE_ASYNC
        async_b128(la,     ap);
        async_b128(la + 8, ap + 8);
        async_b128(lb,     bp);
#else
        *(uint4*)la       = *(const uint4*)ap;
        *(uint4*)(la + 8) = *(const uint4*)(ap + 8);
        *(uint4*)lb       = *(const uint4*)bp;
#endif
    };

    stage(0, 0);
    async_wait0();
    __syncthreads();

    int p = 0;
    for (int k0 = 0; k0 < Kdim; k0 += 32) {
        if (k0 + 32 < Kdim) stage(p ^ 1, k0 + 32);   // prefetch next tile

        FragB fa[2], fb[2];
#pragma unroll
        for (int s = 0; s < 2; ++s) {
            const int arow = wm * 32 + s * 16 + lr;
            fa[s].u[0] = *(const uint4*)(&As[p][arow * 32 + lh * 8]);
            fa[s].u[1] = *(const uint4*)(&As[p][arow * 32 + 16 + lh * 8]);
            const int bcol = wn * 32 + s * 16 + lr;
            fb[s].u[0] = *(const uint4*)(&Bs[p][bcol * 32 + lh * 8]);
            fb[s].u[1] = *(const uint4*)(&Bs[p][bcol * 32 + 16 + lh * 8]);
        }
#pragma unroll
        for (int i = 0; i < 2; ++i)
#pragma unroll
            for (int j = 0; j < 2; ++j)
                acc[i][j] = wmma_bf16(fa[i].v, fb[j].v, acc[i][j]);

        async_wait0();        // next tile landed in LDS (own wave's asyncs)
        __syncthreads();      // everyone done reading buffer p
        p ^= 1;
    }

#pragma unroll
    for (int i = 0; i < 2; ++i)
#pragma unroll
        for (int j = 0; j < 2; ++j)
#pragma unroll
            for (int e = 0; e < 8; ++e) {
                const int row = m0 + wm * 32 + i * 16 + lh * 8 + e;
                const int col = n0 + wn * 32 + j * 16 + lr;
                const float val = acc[i][j][e] + bias[col];
                if (MODE == 0) {
                    const int bb = row >> 11, ss = row & 2047;
                    const int hh = col >> 6,  dd = col & 63;
                    ((__bf16*)outv)[((size_t)(bb * Hd + hh) * Sd + ss) * Dd + dd] =
                        (__bf16)val;
                } else {
                    ((float*)outv)[(size_t)row * Ndim + col] = val;
                }
            }
}

// ---------------------------------------------------------------------------
// Flash attention: per (b,h) and 64-query block. 128 threads = 4 waves;
// each wave owns 16 query rows and the full D=64 output.
// ---------------------------------------------------------------------------
__global__ __launch_bounds__(128) void attn_flash(
    const __bf16* __restrict__ Q, const __bf16* __restrict__ K,
    const __bf16* __restrict__ V, __bf16* __restrict__ Y)
{
    __shared__ __bf16 Ks[32 * 64];      // [key][d]       4 KB
    __shared__ __bf16 Vt[64 * 32];      // [d][key]       4 KB
    __shared__ __bf16 Ps[4][16 * 32];   // per-wave P     4 KB

    const int t    = threadIdx.x;
    const int lane = t & 31;
    const int w    = t >> 5;
    const int lh   = lane >> 4;
    const int lr   = lane & 15;
    const int bh   = blockIdx.y;            // 0..31
    const int b    = bh >> 4;
    const int h    = bh & 15;
    const int q0   = blockIdx.x * 64;
    const size_t hoff = (size_t)bh * Sd * Dd;
    const __bf16* Qh = Q + hoff;
    const __bf16* Kh = K + hoff;
    const __bf16* Vh = V + hoff;
    const int qbase = q0 + w * 16;

    // Q A-fragments for the two 32-wide D chunks (register resident).
    FragB qf[2];
#pragma unroll
    for (int c = 0; c < 2; ++c) {
        const __bf16* qp = Qh + (size_t)(qbase + lr) * Dd + c * 32 + lh * 8;
        qf[c].u[0] = *(const uint4*)qp;
        qf[c].u[1] = *(const uint4*)(qp + 16);
    }

    f32x8 acc_o[4] = {};
    float mrun[8], lsum[8];
#pragma unroll
    for (int i = 0; i < 8; ++i) { mrun[i] = -1e30f; lsum[i] = 0.0f; }

    const int vr = t >> 2;          // 0..31 (key row)
    const int vc = (t & 3) * 16;    // 0..48 (d base)

    for (int kb = 0; kb < Sd; kb += 32) {
        // stage K block [32][64] (async global->LDS when available)
        const __bf16* kp = Kh + (size_t)(kb + vr) * Dd + vc;
#if USE_ASYNC
        async_b128(Ks + vr * 64 + vc,     kp);
        async_b128(Ks + vr * 64 + vc + 8, kp + 8);
#else
        *(uint4*)(Ks + vr * 64 + vc)     = *(const uint4*)kp;
        *(uint4*)(Ks + vr * 64 + vc + 8) = *(const uint4*)(kp + 8);
#endif
        // stage V block transposed -> Vt[d][key] (needs VGPR round-trip)
        union { uint4 u[2]; __bf16 hh[16]; } vtmp;
        const __bf16* vp = Vh + (size_t)(kb + vr) * Dd + vc;
        vtmp.u[0] = *(const uint4*)vp;
        vtmp.u[1] = *(const uint4*)(vp + 8);
#pragma unroll
        for (int j = 0; j < 16; ++j) Vt[(vc + j) * 32 + vr] = vtmp.hh[j];
        async_wait0();
        __syncthreads();

        // S = Q * K^T  (two 16x16 tiles of keys, 2 K-chunks each)
        f32x8 sc[2] = {};
#pragma unroll
        for (int nt = 0; nt < 2; ++nt) {
#pragma unroll
            for (int c = 0; c < 2; ++c) {
                FragB fb;
                const __bf16* bp = Ks + (nt * 16 + lr) * 64 + c * 32 + lh * 8;
                fb.u[0] = *(const uint4*)bp;
                fb.u[1] = *(const uint4*)(bp + 16);
                sc[nt] = wmma_bf16(qf[c].v, fb.v, sc[nt]);
            }
        }

        // online softmax (row = lh*8 + i, 16-lane reductions)
#pragma unroll
        for (int i = 0; i < 8; ++i) {
            float s0 = sc[0][i] * 0.125f;   // 1/sqrt(64)
            float s1 = sc[1][i] * 0.125f;
            float mx = fmaxf(s0, s1);
#pragma unroll
            for (int m = 1; m < 16; m <<= 1) mx = fmaxf(mx, __shfl_xor(mx, m, 32));
            const float mnew = fmaxf(mrun[i], mx);
            const float p0 = __expf(s0 - mnew);
            const float p1 = __expf(s1 - mnew);
            float rs = p0 + p1;
#pragma unroll
            for (int m = 1; m < 16; m <<= 1) rs += __shfl_xor(rs, m, 32);
            const float alpha = __expf(mrun[i] - mnew);
            lsum[i] = lsum[i] * alpha + rs;
            mrun[i] = mnew;
#pragma unroll
            for (int n = 0; n < 4; ++n) acc_o[n][i] *= alpha;
            const int mrow = lh * 8 + i;
            Ps[w][mrow * 32 + lr]      = (__bf16)p0;
            Ps[w][mrow * 32 + 16 + lr] = (__bf16)p1;
        }

        // O += P * V  (P reloaded from LDS in A-layout; same-wave DS is in-order)
        FragB fp;
        fp.u[0] = *(const uint4*)(&Ps[w][lr * 32 + lh * 8]);
        fp.u[1] = *(const uint4*)(&Ps[w][lr * 32 + 16 + lh * 8]);
#pragma unroll
        for (int n = 0; n < 4; ++n) {
            FragB fv;
            const __bf16* bp = Vt + (n * 16 + lr) * 32 + lh * 8;
            fv.u[0] = *(const uint4*)bp;
            fv.u[1] = *(const uint4*)(bp + 16);
            acc_o[n] = wmma_bf16(fp.v, fv.v, acc_o[n]);
        }
        __syncthreads();
    }

    // normalize and scatter to [B,S,E] bf16 (E-major so it feeds the proj GEMM)
#pragma unroll
    for (int n = 0; n < 4; ++n)
#pragma unroll
        for (int i = 0; i < 8; ++i) {
            const int q = qbase + lh * 8 + i;
            const int d = n * 16 + lr;
            const float val = acc_o[n][i] / lsum[i];
            Y[(size_t)(b * Sd + q) * Ed + h * Dd + d] = (__bf16)val;
        }
}

// ---------------------------------------------------------------------------
// LayerNorm (ddof=1, matches torch (x-mean)/(std+eps)*gain+beta), E=1024.
// ---------------------------------------------------------------------------
__global__ __launch_bounds__(256) void layernorm_k(
    const float* __restrict__ X, const float* __restrict__ gain,
    const float* __restrict__ beta, float* __restrict__ out)
{
    __shared__ float red[16];
    const int row = blockIdx.x;
    const int t   = threadIdx.x;
    const float* xr = X + (size_t)row * Ed;
    float4 v = *(const float4*)(xr + t * 4);
    float sum = v.x + v.y + v.z + v.w;
    float sq  = v.x * v.x + v.y * v.y + v.z * v.z + v.w * v.w;
#pragma unroll
    for (int m = 1; m < 32; m <<= 1) {
        sum += __shfl_xor(sum, m, 32);
        sq  += __shfl_xor(sq,  m, 32);
    }
    if ((t & 31) == 0) { red[t >> 5] = sum; red[8 + (t >> 5)] = sq; }
    __syncthreads();
    float sa = 0.0f, qa = 0.0f;
#pragma unroll
    for (int i = 0; i < 8; ++i) { sa += red[i]; qa += red[8 + i]; }
    const float mean = sa * (1.0f / 1024.0f);
    const float var  = fmaxf(qa - 1024.0f * mean * mean, 0.0f) * (1.0f / 1023.0f);
    const float inv  = 1.0f / (sqrtf(var) + 1e-6f);
    float4 g  = *(const float4*)(gain + t * 4);
    float4 bt = *(const float4*)(beta + t * 4);
    float4 o;
    o.x = (v.x - mean) * inv * g.x + bt.x;
    o.y = (v.y - mean) * inv * g.y + bt.y;
    o.z = (v.z - mean) * inv * g.z + bt.z;
    o.w = (v.w - mean) * inv * g.w + bt.w;
    *(float4*)(out + (size_t)row * Ed + t * 4) = o;
}

// ---------------------------------------------------------------------------
// Host launcher
// ---------------------------------------------------------------------------
extern "C" void kernel_launch(void* const* d_in, const int* in_sizes, int n_in,
                              void* d_out, int out_size, void* d_ws, size_t ws_size,
                              hipStream_t stream)
{
    (void)in_sizes; (void)n_in; (void)out_size; (void)ws_size;

    const float* x    = (const float*)d_in[0];
    const float* Wq   = (const float*)d_in[1];
    const float* bq   = (const float*)d_in[2];
    const float* Wk   = (const float*)d_in[3];
    const float* bk   = (const float*)d_in[4];
    const float* Wv   = (const float*)d_in[5];
    const float* bv   = (const float*)d_in[6];
    const float* Wp   = (const float*)d_in[7];
    const float* bp   = (const float*)d_in[8];
    const float* gain = (const float*)d_in[9];
    const float* beta = (const float*)d_in[10];

    char* ws = (char*)d_ws;
    size_t off = 0;
    auto carve = [&](size_t bytes) -> void* {
        void* p = ws + off;
        off += (bytes + 255) & ~(size_t)255;
        return p;
    };
    __bf16* xbf  = (__bf16*)carve((size_t)Md * Ed * 2);
    __bf16* wqt  = (__bf16*)carve((size_t)Ed * Ed * 2);
    __bf16* wkt  = (__bf16*)carve((size_t)Ed * Ed * 2);
    __bf16* wvt  = (__bf16*)carve((size_t)Ed * Ed * 2);
    __bf16* wpt  = (__bf16*)carve((size_t)Ed * Ed * 2);
    __bf16* qbuf = (__bf16*)carve((size_t)Md * Ed * 2);
    __bf16* kbuf = (__bf16*)carve((size_t)Md * Ed * 2);
    __bf16* vbuf = (__bf16*)carve((size_t)Md * Ed * 2);
    __bf16* ybf  = (__bf16*)carve((size_t)Md * Ed * 2);
    float*  proj = (float*) carve((size_t)Md * Ed * 4);

    // 1) conversions
    cvt_f32_bf16<<<(Md * Ed / 4) / 256, 256, 0, stream>>>(x, xbf, Md * Ed);
    transpose_cvt<<<(Ed * Ed) / 256, 256, 0, stream>>>(Wq, wqt);
    transpose_cvt<<<(Ed * Ed) / 256, 256, 0, stream>>>(Wk, wkt);
    transpose_cvt<<<(Ed * Ed) / 256, 256, 0, stream>>>(Wv, wvt);
    transpose_cvt<<<(Ed * Ed) / 256, 256, 0, stream>>>(Wp, wpt);

    // 2) QKV projections (head-scatter epilogue)
    dim3 gg(Ed / 64, Md / 128);
    gemm_bf16<0><<<gg, 256, 0, stream>>>(xbf, wqt, bq, qbuf, Ed, Ed);
    gemm_bf16<0><<<gg, 256, 0, stream>>>(xbf, wkt, bk, kbuf, Ed, Ed);
    gemm_bf16<0><<<gg, 256, 0, stream>>>(xbf, wvt, bv, vbuf, Ed, Ed);

    // 3) flash attention
    attn_flash<<<dim3(Sd / 64, Bd * Hd), 128, 0, stream>>>(qbuf, kbuf, vbuf, ybf);

    // 4) output projection (f32 epilogue)
    gemm_bf16<1><<<gg, 256, 0, stream>>>(ybf, wpt, bp, proj, Ed, Ed);

    // 5) layernorm -> d_out
    layernorm_k<<<Md, 256, 0, stream>>>(proj, gain, beta, (float*)d_out);
}
